// RNNQNetwork_67078799229411
// MI455X (gfx1250) — matrix operand order repaired and checked by reading, not compiled, not run on hardware
//
#include <hip/hip_runtime.h>
#include <hip/hip_bf16.h>

#define Bdim 256
#define Tdim 512
#define Idim 256
#define Hdim 512
#define Adim 18

typedef __attribute__((ext_vector_type(16))) __bf16       v16bf;
typedef __attribute__((ext_vector_type(8)))  float        v8f;
typedef __attribute__((ext_vector_type(8)))  unsigned int v8u;

__device__ __forceinline__ unsigned short f2bf(float f) {
  unsigned int u = __builtin_bit_cast(unsigned int, f);
  u += 0x7FFFu + ((u >> 16) & 1u);   // round-to-nearest-even
  return (unsigned short)(u >> 16);
}

// A fragment (16x32 bf16), src row-major [16][stride] of bf16 bit patterns.
// ISA 7.12.2: lane l<16 -> row M=l, VGPR v holds K = (v&3)*2 + (v>>2)*16 (+1);
// lanes 16-31 same rows, K offset +8. bf16 pairs are contiguous -> dword loads.
__device__ __forceinline__ v16bf frag_a(const unsigned short* src, int stride,
                                        int k0, int lane) {
  int l = lane & 15, half = lane >> 4;
  const unsigned int* s32 = (const unsigned int*)src;
  v8u u;
#pragma unroll
  for (int v = 0; v < 8; ++v) {
    int kb = ((v & 3) << 1) + (half << 3) + ((v >> 2) << 4);
    u[v] = s32[(l * stride + k0 + kb) >> 1];
  }
  return __builtin_bit_cast(v16bf, u);
}

// B fragment (32x16 bf16): B[k][n] = src[n*stride + k] (src row-major [N][K]).
// lane l: N=l&15, VGPR v holds K = (v&3)*2 + (v>>2)*16 (+1); lanes 16-31 K offset +8.
__device__ __forceinline__ v16bf frag_b(const unsigned short* src, int stride,
                                        int k0, int lane) {
  int l = lane & 15, half = lane >> 4;
  const unsigned int* s32 = (const unsigned int*)src;
  v8u u;
#pragma unroll
  for (int v = 0; v < 8; ++v) {
    int k = ((v & 3) << 1) + (half << 3) + ((v >> 2) << 4);
    u[v] = s32[(l * stride + k0 + k) >> 1];
  }
  return __builtin_bit_cast(v16bf, u);
}

// ---------- prep: weights -> bf16 scratch, copy lengths bits to d_out tail ----
__global__ void __launch_bounds__(256) prep_kernel(
    const float* __restrict__ W_ih, const float* __restrict__ W_hh,
    const float* __restrict__ W_fc, const int* __restrict__ lengths,
    unsigned short* __restrict__ Wih_b, unsigned short* __restrict__ Whh_b,
    unsigned short* __restrict__ Wfc_b, int* __restrict__ len_out) {
  int i = blockIdx.x * blockDim.x + threadIdx.x;
  const int nIh = Hdim * Idim;          // 131072
  const int nHh = Hdim * Hdim;          // 262144
  const int nFc = 32 * Hdim;            // 16384 (A padded 18 -> 32)
  int total = nIh + nHh + nFc;
  for (int e = i; e < total; e += gridDim.x * blockDim.x) {
    if (e < nIh) {
      Wih_b[e] = f2bf(W_ih[e]);
    } else if (e < nIh + nHh) {
      int e2 = e - nIh;
      Whh_b[e2] = f2bf(W_hh[e2]);
    } else {
      int e2 = e - nIh - nHh;
      int a = e2 / Hdim;
      Wfc_b[e2] = (a < Adim) ? f2bf(W_fc[e2]) : (unsigned short)0;
    }
  }
  if (i < Bdim) len_out[i] = lengths[i];
}

// ---------- proj GEMM: proj[b*T+t][h] = x @ W_ih^T + b_ih + b_hh -------------
__global__ void __launch_bounds__(256) proj_gemm(
    const float* __restrict__ x, const float* __restrict__ b_ih,
    const float* __restrict__ b_hh, const unsigned short* __restrict__ Wih,
    float* __restrict__ proj) {
  __shared__ __align__(16) unsigned short xs[16 * Idim];  // 8 KB bf16 A-strip
  int m0 = blockIdx.x * 16;
  int tid = threadIdx.x;
  for (int e = tid; e < 16 * Idim; e += 256)              // coalesced fp32 read
    xs[e] = f2bf(x[(size_t)m0 * Idim + e]);
  __syncthreads();

  int lane = tid & 31, wave = tid >> 5;
  int l = lane & 15, half = lane >> 4;
#pragma unroll
  for (int q = 0; q < 4; ++q) {                           // 8 waves x 4 = 32 N-tiles
    int n0 = (wave * 4 + q) * 16;
    v8f acc = {};
#pragma unroll
    for (int k0 = 0; k0 < Idim; k0 += 32) {
      v16bf a = frag_a(xs, Idim, k0, lane);
      v16bf b = frag_b(Wih + (size_t)n0 * Idim, Idim, k0, lane);
      acc = __builtin_amdgcn_wmma_f32_16x16x32_bf16(false, a, false, b,
                                                    (short)0, acc, false, false);
    }
    int ng = n0 + l;
    float bias = b_ih[ng] + b_hh[ng];
#pragma unroll
    for (int r = 0; r < 8; ++r) {
      int m = m0 + r + half * 8;
      proj[(size_t)m * Hdim + ng] = acc[r] + bias;
    }
  }
}

// ---------- persistent recurrence + fused FC head -----------------------------
// One workgroup owns 16 batch rows; no inter-WG deps (batch rows independent).
__global__ void __launch_bounds__(256) rnn_persist(
    const float* __restrict__ h0, const int* __restrict__ lengths,
    const float* __restrict__ b_fc, const unsigned short* __restrict__ Whh,
    const unsigned short* __restrict__ Wfc, const float* __restrict__ proj,
    float* __restrict__ out) {
  __shared__ __align__(16) float          hS[16 * Hdim];  // 32 KB fp32 carried h
  __shared__ __align__(16) unsigned short hB[16 * Hdim];  // 16 KB bf16 staging
  __shared__ int lenS[16];

  int b0 = blockIdx.x * 16;
  int tid = threadIdx.x;
  for (int e = tid; e < 16 * Hdim; e += 256)
    hS[e] = h0[(size_t)b0 * Hdim + e];                    // h0 is [1,B,H]
  if (tid < 16) lenS[tid] = lengths[b0 + tid];
  __syncthreads();

  int lane = tid & 31, wave = tid >> 5;
  int l = lane & 15, half = lane >> 4;

  for (int t = 0; t < Tdim; ++t) {
    // stage 1: bf16 copy of carried hidden state (unmasked)
    for (int e = tid; e < 16 * Hdim; e += 256) hB[e] = f2bf(hS[e]);
    __syncthreads();

    // stage 2: h_new = tanh(proj_t + h @ W_hh^T); masked state update
#pragma unroll
    for (int q = 0; q < 4; ++q) {
      int n0 = (wave * 4 + q) * 16;
      int ng = n0 + l;
      v8f acc;
#pragma unroll
      for (int r = 0; r < 8; ++r) {
        int m = r + half * 8;
        acc[r] = proj[((size_t)(b0 + m) * Tdim + t) * Hdim + ng];
      }
#pragma unroll
      for (int k0 = 0; k0 < Hdim; k0 += 32) {
        v16bf a = frag_a(hB, Hdim, k0, lane);
        v16bf b = frag_b(Whh + (size_t)n0 * Hdim, Hdim, k0, lane);
        acc = __builtin_amdgcn_wmma_f32_16x16x32_bf16(false, a, false, b,
                                                      (short)0, acc, false, false);
      }
#pragma unroll
      for (int r = 0; r < 8; ++r) {
        int m = r + half * 8;
        if (t < lenS[m]) hS[m * Hdim + ng] = tanhf(acc[r]);
      }
    }
    __syncthreads();

    // stage 3: masked bf16 output staging (zero past sequence length)
    for (int e = tid; e < 16 * Hdim; e += 256) {
      int m = e >> 9;
      hB[e] = (t < lenS[m]) ? f2bf(hS[e]) : (unsigned short)0;
    }
    __syncthreads();

    // stage 4: FC head via WMMA, A padded to 32 -> 2 N-tiles on waves 0..1
    if (wave < 2) {
      int n0 = wave * 16;
      v8f acc = {};
#pragma unroll
      for (int k0 = 0; k0 < Hdim; k0 += 32) {
        v16bf a = frag_a(hB, Hdim, k0, lane);
        v16bf b = frag_b(Wfc + (size_t)n0 * Hdim, Hdim, k0, lane);
        acc = __builtin_amdgcn_wmma_f32_16x16x32_bf16(false, a, false, b,
                                                      (short)0, acc, false, false);
      }
      int ai = n0 + l;
      if (ai < Adim) {
        float bias = b_fc[ai];
#pragma unroll
        for (int r = 0; r < 8; ++r) {
          int m = r + half * 8;
          out[((size_t)(b0 + m) * Tdim + t) * Adim + ai] = acc[r] + bias;
        }
      }
    }
    __syncthreads();   // protect hB before next iteration's stage 1
  }
}

extern "C" void kernel_launch(void* const* d_in, const int* in_sizes, int n_in,
                              void* d_out, int out_size, void* d_ws, size_t ws_size,
                              hipStream_t stream) {
  const float* x       = (const float*)d_in[0];
  const float* h0      = (const float*)d_in[1];
  const int*   lengths = (const int*)d_in[2];
  const float* W_ih    = (const float*)d_in[3];
  const float* W_hh    = (const float*)d_in[4];
  const float* b_ih    = (const float*)d_in[5];
  const float* b_hh    = (const float*)d_in[6];
  const float* W_fc    = (const float*)d_in[7];
  const float* b_fc    = (const float*)d_in[8];
  float* out = (float*)d_out;

  char* ws = (char*)d_ws;
  unsigned short* Wih_b = (unsigned short*)(ws);            // 256 KB
  unsigned short* Whh_b = (unsigned short*)(ws + 262144);   // 512 KB
  unsigned short* Wfc_b = (unsigned short*)(ws + 786432);   //  32 KB (padded 32xH)
  float*          proj  = (float*)(ws + 1048576);           // 256 MB [B*T, H]
  int* len_out = (int*)d_out + (size_t)Bdim * Tdim * Adim;  // lengths bits in tail

  prep_kernel<<<512, 256, 0, stream>>>(W_ih, W_hh, W_fc, lengths,
                                       Wih_b, Whh_b, Wfc_b, len_out);
  proj_gemm<<<(Bdim * Tdim) / 16, 256, 0, stream>>>(x, b_ih, b_hh, Wih_b, proj);
  rnn_persist<<<Bdim / 16, 256, 0, stream>>>(h0, lengths, b_fc, Whh_b, Wfc_b,
                                             proj, out);
}
